// TerraTNTEnv_69483980915123
// MI455X (gfx1250) — compile-verified, matrix operand order checked by reading, not run on hardware
//
#include <hip/hip_runtime.h>
#include <hip/hip_bf16.h>
#include <math.h>

#define HALFW 70.0f   // 140.0 * 0.5
#define FUT 360

typedef _Float16 v16h __attribute__((ext_vector_type(16)));
typedef _Float16 v8h  __attribute__((ext_vector_type(8)));
typedef float    v8f  __attribute__((ext_vector_type(8)));

__device__ __forceinline__ float sigm_(float x) { return 1.0f / (1.0f + __expf(-x)); }

// ---------------------------------------------------------------------------
// f16 fragment load for 16x16x32 WMMA, K-major rows, K padded to 32 multiple.
// ISA layout: element e<8 -> K = 8*(lane>=16) + e ; e>=8 -> K = 16 + 8*(lane>=16) + (e-8)
// => per lane two contiguous 16-byte loads.
// p = row_base + kbase (16B aligned: rows and kbase are multiples of 32 halves)
// ---------------------------------------------------------------------------
__device__ __forceinline__ v16h ld_frag16(const _Float16* __restrict__ p, int lane) {
    const _Float16* q = p + ((lane >> 4) << 3);
    v8h lo = *(const v8h*)(q);
    v8h hi = *(const v8h*)(q + 16);
    v16h f;
#pragma unroll
    for (int e = 0; e < 8; ++e) { f[e] = lo[e]; f[e + 8] = hi[e]; }
    return f;
}

__device__ __forceinline__ v16h zero_frag16() {
    v16h f;
#pragma unroll
    for (int e = 0; e < 16; ++e) f[e] = (_Float16)0.0f;
    return f;
}

#define WMMA(A, B, C) __builtin_amdgcn_wmma_f32_16x16x32_f16(false, A, false, B, (short)0, C, false, false)

// ---------------------------------------------------------------------------
// Conversion / re-layout pre-passes (tiny, run once per launch)
// ---------------------------------------------------------------------------
// (R,K) f32 -> (R,Kp) f16 zero-padded
__global__ void cvt_pad_kernel(const float* __restrict__ in, _Float16* __restrict__ out,
                               int K, int Kp, int total) {
    int idx = blockIdx.x * 256 + threadIdx.x;
    if (idx >= total) return;
    int r = idx / Kp, j = idx - r * Kp;
    out[idx] = (j < K) ? (_Float16)in[(size_t)r * K + j] : (_Float16)0.0f;
}

// conv weights (O, I, 3, 3) f32 -> (O, 9, I) f16
__global__ void conv_w_reorder_kernel(const float* __restrict__ in, _Float16* __restrict__ out,
                                      int I, int total) {
    int idx = blockIdx.x * 256 + threadIdx.x;
    if (idx >= total) return;
    int o = idx / (9 * I);
    int rem = idx - o * 9 * I;
    int rs = rem / I, i = rem - rs * I;
    out[idx] = (_Float16)in[((size_t)o * I + i) * 9 + rs];
}

// ---------------------------------------------------------------------------
// Conv1: A gathered from f32 NCHW input (Cin=18), B fast f16 (K padded 162->192).
// Output NHWC f16 + ReLU.  grid = (ceil(OH*OW/16), Cout/16, B), 32 thr
// ---------------------------------------------------------------------------
template <int Cin, int Hin, int Win, int Cout, int Hout, int Wout, int Kp>
__global__ __launch_bounds__(32) void conv_gather_kernel(
    const float* __restrict__ in, const _Float16* __restrict__ wgt,
    const float* __restrict__ bias, _Float16* __restrict__ out) {
    constexpr int K = Cin * 9;
    constexpr int HW = Hout * Wout;
    int lane = threadIdx.x;
    int mtile = blockIdx.x, ntile = blockIdx.y, b = blockIdx.z;
    int mIdx = mtile * 16 + (lane & 15);
    bool mvalid = mIdx < HW;
    int oh = mIdx / Wout, ow = mIdx % Wout;
    int n = ntile * 16 + (lane & 15);
    const float* inb = in + (size_t)b * Cin * Hin * Win;
    const _Float16* wr = wgt + (size_t)n * Kp;

    v8f acc = {};
    for (int kb = 0; kb < Kp; kb += 32) {
        v16h a;
#pragma unroll
        for (int e = 0; e < 16; ++e) {
            int v = e >> 1, h = e & 1;
            int k = kb + ((v & 4) << 2) + ((lane >> 4) << 3) + ((v & 3) << 1) + h;
            float av = 0.0f;
            if (mvalid && k < K) {
                int cin = k / 9, rs = k % 9;
                int r = rs / 3, s = rs % 3;
                int ih = oh * 2 - 1 + r, iw = ow * 2 - 1 + s;
                if ((unsigned)ih < (unsigned)Hin && (unsigned)iw < (unsigned)Win)
                    av = inb[((size_t)cin * Hin + ih) * Win + iw];
            }
            a[e] = (_Float16)av;
        }
        v16h bf = ld_frag16(wr + kb, lane);
        acc = WMMA(a, bf, acc);
    }
    float bv = bias[n];
    _Float16* outb = out + (size_t)b * HW * Cout;
#pragma unroll
    for (int d = 0; d < 8; ++d) {
        int m = mtile * 16 + d + ((lane >> 4) << 3);
        if (m < HW) {
            float v = acc[d] + bv;
            outb[(size_t)m * Cout + n] = (_Float16)(v > 0.0f ? v : 0.0f);
        }
    }
}

// ---------------------------------------------------------------------------
// Convs 2-4: NHWC f16 in/out, weights (Cout, 9, Cin) f16, Cin % 32 == 0 so a
// 32-chunk sits inside one (r,s) tap => A fragment is two b128 loads.
// ---------------------------------------------------------------------------
template <int Cin, int Hin, int Win, int Cout, int Hout, int Wout>
__global__ __launch_bounds__(32) void conv_nhwc_kernel(
    const _Float16* __restrict__ in, const _Float16* __restrict__ wgt,
    const float* __restrict__ bias, _Float16* __restrict__ out) {
    constexpr int K = Cin * 9;
    constexpr int HW = Hout * Wout;
    int lane = threadIdx.x;
    int mtile = blockIdx.x, ntile = blockIdx.y, b = blockIdx.z;
    int mIdx = mtile * 16 + (lane & 15);
    bool mvalid = mIdx < HW;
    int oh = mIdx / Wout, ow = mIdx % Wout;
    int n = ntile * 16 + (lane & 15);
    const _Float16* inb = in + (size_t)b * Hin * Win * Cin;
    const _Float16* wr = wgt + (size_t)n * K;

    v8f acc = {};
    for (int kb = 0; kb < K; kb += 32) {
        int rs = kb / Cin;
        int r = rs / 3, s = rs % 3;
        int cbase = kb - rs * Cin;
        int ih = oh * 2 - 1 + r, iw = ow * 2 - 1 + s;
        v16h a;
        if (mvalid && (unsigned)ih < (unsigned)Hin && (unsigned)iw < (unsigned)Win)
            a = ld_frag16(inb + ((size_t)ih * Win + iw) * Cin + cbase, lane);
        else
            a = zero_frag16();
        __builtin_prefetch(wr + kb + 32, 0, 1);
        v16h bf = ld_frag16(wr + kb, lane);
        acc = WMMA(a, bf, acc);
    }
    float bv = bias[n];
    _Float16* outb = out + (size_t)b * HW * Cout;
#pragma unroll
    for (int d = 0; d < 8; ++d) {
        int m = mtile * 16 + d + ((lane >> 4) << 3);
        if (m < HW) {
            float v = acc[d] + bv;
            outb[(size_t)m * Cout + n] = (_Float16)(v > 0.0f ? v : 0.0f);
        }
    }
}

// ---------------------------------------------------------------------------
// Fused LSTM step: gates = X@Wih^T + Hprev@Whh^T + b ; cell update.
// All f16 operands, K multiples of 32, B=64, grid = (4, H/16), wave/block.
// h written f16, c kept f32.
// ---------------------------------------------------------------------------
__global__ __launch_bounds__(32) void lstm_step_kernel(
    const _Float16* __restrict__ X, int ldx, int Kx,
    const _Float16* __restrict__ Wih,
    const _Float16* __restrict__ Hprev, const _Float16* __restrict__ Whh, int H,
    const float* __restrict__ bias, const float* __restrict__ Cprev,
    _Float16* __restrict__ Hout, float* __restrict__ Cout) {
    int lane = threadIdx.x;
    int mb = blockIdx.x * 16, nb = blockIdx.y * 16;
    int mrow = mb + (lane & 15);
    int ncol = nb + (lane & 15);

    const _Float16* xr = X + (size_t)mrow * ldx;
    const _Float16* w0 = Wih + (size_t)(0 * H + ncol) * Kx;
    const _Float16* w1 = Wih + (size_t)(1 * H + ncol) * Kx;
    const _Float16* w2 = Wih + (size_t)(2 * H + ncol) * Kx;
    const _Float16* w3 = Wih + (size_t)(3 * H + ncol) * Kx;

    v8f a0 = {}, a1 = {}, a2 = {}, a3 = {};
    for (int kb = 0; kb < Kx; kb += 32) {
        v16h a = ld_frag16(xr + kb, lane);
        __builtin_prefetch(w0 + kb + 32, 0, 1);
        a0 = WMMA(a, ld_frag16(w0 + kb, lane), a0);
        a1 = WMMA(a, ld_frag16(w1 + kb, lane), a1);
        a2 = WMMA(a, ld_frag16(w2 + kb, lane), a2);
        a3 = WMMA(a, ld_frag16(w3 + kb, lane), a3);
    }
    const _Float16* hr = Hprev + (size_t)mrow * H;
    const _Float16* u0 = Whh + (size_t)(0 * H + ncol) * H;
    const _Float16* u1 = Whh + (size_t)(1 * H + ncol) * H;
    const _Float16* u2 = Whh + (size_t)(2 * H + ncol) * H;
    const _Float16* u3 = Whh + (size_t)(3 * H + ncol) * H;
    for (int kb = 0; kb < H; kb += 32) {
        v16h a = ld_frag16(hr + kb, lane);
        __builtin_prefetch(u0 + kb + 32, 0, 1);
        a0 = WMMA(a, ld_frag16(u0 + kb, lane), a0);
        a1 = WMMA(a, ld_frag16(u1 + kb, lane), a1);
        a2 = WMMA(a, ld_frag16(u2 + kb, lane), a2);
        a3 = WMMA(a, ld_frag16(u3 + kb, lane), a3);
    }
    float bi = bias[0 * H + ncol], bf = bias[1 * H + ncol];
    float bg = bias[2 * H + ncol], bo = bias[3 * H + ncol];
#pragma unroll
    for (int d = 0; d < 8; ++d) {
        int m = mb + d + ((lane >> 4) << 3);
        size_t idx = (size_t)m * H + ncol;
        float ig = sigm_(a0[d] + bi);
        float fg = sigm_(a1[d] + bf);
        float gg = tanhf(a2[d] + bg);
        float og = sigm_(a3[d] + bo);
        float c = fg * Cprev[idx] + ig * gg;
        Cout[idx] = c;
        Hout[idx] = (_Float16)(og * tanhf(c));
    }
}

// ---------------------------------------------------------------------------
// Generic GEMM: C(f16) = act([A1|A2] @ W^T + bias) (+ add_f32)
// W is (N, K1+K2) f16 row-major, K1,K2 multiples of 32.
// grid = (M/16, N/16), wave per block.
// ---------------------------------------------------------------------------
__global__ __launch_bounds__(32) void gemm_ba_kernel(
    const _Float16* __restrict__ A1, int ld1, int K1,
    const _Float16* __restrict__ A2, int ld2, int K2,
    const _Float16* __restrict__ W, const float* __restrict__ bias,
    const float* __restrict__ add, _Float16* __restrict__ C, int ldc, int do_relu) {
    int lane = threadIdx.x;
    int mb = blockIdx.x * 16, nb = blockIdx.y * 16;
    int mrow = mb + (lane & 15);
    int ncol = nb + (lane & 15);
    int ldw = K1 + K2;
    const _Float16* wr = W + (size_t)ncol * ldw;

    v8f acc = {};
    for (int kb = 0; kb < K1; kb += 32) {
        v16h a = ld_frag16(A1 + (size_t)mrow * ld1 + kb, lane);
        acc = WMMA(a, ld_frag16(wr + kb, lane), acc);
    }
    for (int kb = 0; kb < K2; kb += 32) {
        v16h a = ld_frag16(A2 + (size_t)mrow * ld2 + kb, lane);
        acc = WMMA(a, ld_frag16(wr + K1 + kb, lane), acc);
    }
    float bv = bias[ncol];
#pragma unroll
    for (int d = 0; d < 8; ++d) {
        int m = mb + d + ((lane >> 4) << 3);
        float v = acc[d] + bv;
        if (do_relu) v = v > 0.0f ? v : 0.0f;
        if (add) v += add[(size_t)m * ldc + ncol];
        C[(size_t)m * ldc + ncol] = (_Float16)v;
    }
}

// ---------------------------------------------------------------------------
// Mean pool over 14x14 (NHWC f16).  grid = 64, block = 128
// ---------------------------------------------------------------------------
__global__ void pool_kernel(const _Float16* __restrict__ es, _Float16* __restrict__ pooled) {
    int b = blockIdx.x, c = threadIdx.x;
    const _Float16* p = es + (size_t)b * 196 * 128 + c;
    float s = 0.0f;
    for (int i = 0; i < 196; ++i) s += (float)p[i * 128];
    pooled[b * 128 + c] = (_Float16)(s * (1.0f / 196.0f));
}

// ---------------------------------------------------------------------------
// Motion feature MLP (f32 out for epilogue add).  grid = 64, block = 256
// ---------------------------------------------------------------------------
__global__ void motion_kernel(const float* __restrict__ hist,
                              const float* __restrict__ Wm1, const float* __restrict__ bm1,
                              const float* __restrict__ Wm2, const float* __restrict__ bm2,
                              float* __restrict__ mf) {
    __shared__ float mo[4];
    __shared__ float m1[64];
    int b = blockIdx.x, tid = threadIdx.x;
    if (tid == 0) {
        const float* hb = hist + (size_t)b * 60 * 26;
        float s2 = 0, s3 = 0, s9 = 0, s6 = 0, s7 = 0;
        for (int t = 50; t < 60; ++t) {
            const float* r = hb + t * 26;
            s2 += r[2]; s3 += r[3]; s9 += r[9]; s6 += r[6]; s7 += r[7];
        }
        mo[0] = s2 * 0.1f; mo[1] = s3 * 0.1f; mo[2] = s9 * 0.1f;
        mo[3] = atan2f(s6 * 0.1f, s7 * 0.1f);
    }
    __syncthreads();
    if (tid < 64) {
        float v = bm1[tid];
        for (int j = 0; j < 4; ++j) v += mo[j] * Wm1[tid * 4 + j];
        m1[tid] = v > 0.0f ? v : 0.0f;
    }
    __syncthreads();
    float v = bm2[tid];
    for (int j = 0; j < 64; ++j) v += m1[j] * Wm2[tid * 64 + j];
    mf[(size_t)b * 256 + tid] = v;
}

// ---------------------------------------------------------------------------
// Decoder pre-step: bilinear sample (NHWC f16 map) + positional MLP + fusion,
// writes dec_in f16 (B x 288, cols 258..287 zero).  grid = 64, block = 256
// ---------------------------------------------------------------------------
__global__ void dec_pre_kernel(const _Float16* __restrict__ es, const float* __restrict__ pos,
                               const float* __restrict__ Ws, const float* __restrict__ bs,
                               const float* __restrict__ Wp1, const float* __restrict__ bp1,
                               const float* __restrict__ Wp2, const float* __restrict__ bp2,
                               const _Float16* __restrict__ context,
                               const float* __restrict__ scale_p,
                               _Float16* __restrict__ dec_in) {
    __shared__ float samp[128];
    __shared__ float pf1[32];
    int b = blockIdx.x, tid = threadIdx.x;
    float p0 = pos[b * 2 + 0], p1 = pos[b * 2 + 1];
    if (tid < 128) {
        float gx = fminf(fmaxf(p0 / HALFW, -1.0f), 1.0f);
        float gy = fminf(fmaxf(-p1 / HALFW, -1.0f), 1.0f);
        float x = (gx + 1.0f) * 0.5f * 13.0f;
        float y = (gy + 1.0f) * 0.5f * 13.0f;
        int x0 = (int)fminf(fmaxf(floorf(x), 0.0f), 12.0f);
        int y0 = (int)fminf(fmaxf(floorf(y), 0.0f), 12.0f);
        float wx = x - (float)x0, wy = y - (float)y0;
        const _Float16* fm = es + (size_t)b * 196 * 128 + tid;
        float v00 = (float)fm[(y0 * 14 + x0) * 128];
        float v01 = (float)fm[(y0 * 14 + x0 + 1) * 128];
        float v10 = (float)fm[((y0 + 1) * 14 + x0) * 128];
        float v11 = (float)fm[((y0 + 1) * 14 + x0 + 1) * 128];
        samp[tid] = v00 * (1 - wx) * (1 - wy) + v01 * wx * (1 - wy) +
                    v10 * (1 - wx) * wy + v11 * wx * wy;
    } else if (tid < 160) {
        int j = tid - 128;
        float pn0 = p0 / HALFW, pn1 = p1 / HALFW;
        float v = pn0 * Wp1[j * 2 + 0] + pn1 * Wp1[j * 2 + 1] + bp1[j];
        pf1[j] = v > 0.0f ? v : 0.0f;
    }
    __syncthreads();
    int n = tid;
    float sp = bs[n];
    for (int c = 0; c < 128; ++c) sp += samp[c] * Ws[n * 128 + c];
    sp = sp > 0.0f ? sp : 0.0f;
    float pf = bp2[n];
    for (int j = 0; j < 32; ++j) pf += pf1[j] * Wp2[n * 32 + j];
    float sc = (float)context[(size_t)b * 256 + n] + scale_p[0] * sp + pf;
    _Float16* dr = dec_in + (size_t)b * 288;
    dr[2 + n] = (_Float16)sc;
    if (n < 2) dr[n] = (_Float16)((n == 0) ? p0 : p1);
    if (n < 30) dr[258 + n] = (_Float16)0.0f;
}

// ---------------------------------------------------------------------------
// Output head: delta = h2 @ Wo^T + bo ; pos += delta ; emit.  1 block, 128 thr
// ---------------------------------------------------------------------------
__global__ void pos_out_kernel(const _Float16* __restrict__ h2, const float* __restrict__ Wo,
                               const float* __restrict__ bo, const float* __restrict__ pos_in,
                               float* __restrict__ pos_out, float* __restrict__ out, int t) {
    int tid = threadIdx.x;
    int b = tid >> 1, dim = tid & 1;
    float v = bo[dim];
    const _Float16* h = h2 + (size_t)b * 256;
    const float* w = Wo + (size_t)dim * 256;
    for (int k = 0; k < 256; ++k) v += (float)h[k] * w[k];
    float p = pos_in[b * 2 + dim] + v;
    pos_out[b * 2 + dim] = p;
    out[(size_t)b * (FUT * 2) + t * 2 + dim] = p;
}

// ---------------------------------------------------------------------------
extern "C" void kernel_launch(void* const* d_in, const int* in_sizes, int n_in,
                              void* d_out, int out_size, void* d_ws, size_t ws_size,
                              hipStream_t stream) {
    auto F = [&](int i) { return (const float*)d_in[i]; };
    const float* env_map = F(0);
    const float* history = F(1);
    const float *ck1 = F(2), *cb1 = F(3), *ck2 = F(4), *cb2 = F(5);
    const float *ck3 = F(6), *cb3 = F(7), *ck4 = F(8), *cb4 = F(9);
    const float *Wg = F(10), *bg = F(11);
    const float *hWih0 = F(12), *hWhh0 = F(13), *hb0 = F(14);
    const float *hWih1 = F(15), *hWhh1 = F(16), *hb1 = F(17);
    const float *Wf = F(18), *bf = F(19), *Ws = F(20), *bs = F(21);
    const float* scale_p = F(22);
    const float *Wm1 = F(23), *bm1 = F(24), *Wm2 = F(25), *bm2 = F(26);
    const float *Wp1 = F(27), *bp1 = F(28), *Wp2 = F(29), *bp2 = F(30);
    const float *dWih0 = F(31), *dWhh0 = F(32), *db0 = F(33);
    const float *dWih1 = F(34), *dWhh1 = F(35), *db1 = F(36);
    const float *Wo = F(37), *bo = F(38);
    (void)in_sizes; (void)n_in; (void)out_size; (void)ws_size;

    // --- workspace layout ---
    uint8_t* base = (uint8_t*)d_ws;
    size_t off = 0;
    auto allocB = [&](size_t bytes) -> void* {
        void* p = (void*)(base + off);
        off = (off + bytes + 255) & ~(size_t)255;
        return p;
    };
    auto allocH = [&](size_t n) { return (_Float16*)allocB(n * 2); };
    auto allocF = [&](size_t n) { return (float*)allocB(n * 4); };

    _Float16* x1 = allocH((size_t)64 * 112 * 112 * 32);
    _Float16* x2 = allocH((size_t)64 * 56 * 56 * 64);
    _Float16* x3 = allocH((size_t)64 * 28 * 28 * 128);
    _Float16* es = allocH((size_t)64 * 14 * 14 * 128);
    _Float16* hist16 = allocH((size_t)64 * 60 * 32);
    // f16 weight copies (zero padded K)
    _Float16* wck1 = allocH(32 * 192);
    _Float16* wck2 = allocH(64 * 288);
    _Float16* wck3 = allocH(128 * 576);
    _Float16* wck4 = allocH(128 * 1152);
    _Float16* whih0 = allocH(512 * 32);
    _Float16* whhh0 = allocH(512 * 128);
    _Float16* whih1 = allocH(512 * 128);
    _Float16* whhh1 = allocH(512 * 128);
    _Float16* wWg = allocH(128 * 128);
    _Float16* wWf = allocH(256 * 256);
    _Float16* wdWih0 = allocH(1024 * 288);
    _Float16* wdWhh0 = allocH(1024 * 256);
    _Float16* wdWih1 = allocH(1024 * 256);
    _Float16* wdWhh1 = allocH(1024 * 256);
    // activations / states
    _Float16* pooled = allocH(64 * 128);
    _Float16* envg = allocH(64 * 128);
    float* mfeat = allocF(64 * 256);
    _Float16* context = allocH(64 * 256);
    _Float16* hh1[2] = {allocH(64 * 128), allocH(64 * 128)};
    float*    hc1[2] = {allocF(64 * 128), allocF(64 * 128)};
    _Float16* hh2[2] = {allocH(64 * 128), allocH(64 * 128)};
    float*    hc2[2] = {allocF(64 * 128), allocF(64 * 128)};
    _Float16* dh1[2] = {allocH(64 * 256), allocH(64 * 256)};
    float*    dc1[2] = {allocF(64 * 256), allocF(64 * 256)};
    _Float16* dh2[2] = {allocH(64 * 256), allocH(64 * 256)};
    float*    dc2[2] = {allocF(64 * 256), allocF(64 * 256)};
    _Float16* dec_in = allocH(64 * 288);
    float* posb[2] = {allocF(64 * 2), allocF(64 * 2)};

    // --- weight / input conversion pre-passes ---
    auto cvt = [&](const float* in, _Float16* out, int R, int K, int Kp) {
        int total = R * Kp;
        cvt_pad_kernel<<<(total + 255) / 256, 256, 0, stream>>>(in, out, K, Kp, total);
    };
    cvt(ck1, wck1, 32, 162, 192);                // OIHW flat rows, padded
    conv_w_reorder_kernel<<<(64 * 288 + 255) / 256, 256, 0, stream>>>(ck2, wck2, 32, 64 * 288);
    conv_w_reorder_kernel<<<(128 * 576 + 255) / 256, 256, 0, stream>>>(ck3, wck3, 64, 128 * 576);
    conv_w_reorder_kernel<<<(128 * 1152 + 255) / 256, 256, 0, stream>>>(ck4, wck4, 128, 128 * 1152);
    cvt(hWih0, whih0, 512, 26, 32);
    cvt(hWhh0, whhh0, 512, 128, 128);
    cvt(hWih1, whih1, 512, 128, 128);
    cvt(hWhh1, whhh1, 512, 128, 128);
    cvt(Wg, wWg, 128, 128, 128);
    cvt(Wf, wWf, 256, 256, 256);
    cvt(dWih0, wdWih0, 1024, 258, 288);
    cvt(dWhh0, wdWhh0, 1024, 256, 256);
    cvt(dWih1, wdWih1, 1024, 256, 256);
    cvt(dWhh1, wdWhh1, 1024, 256, 256);
    cvt(history, hist16, 64 * 60, 26, 32);

    // --- CNN encoder ---
    conv_gather_kernel<18, 224, 224, 32, 112, 112, 192>
        <<<dim3(784, 2, 64), 32, 0, stream>>>(env_map, wck1, cb1, x1);
    conv_nhwc_kernel<32, 112, 112, 64, 56, 56>
        <<<dim3(196, 4, 64), 32, 0, stream>>>(x1, wck2, cb2, x2);
    conv_nhwc_kernel<64, 56, 56, 128, 28, 28>
        <<<dim3(49, 8, 64), 32, 0, stream>>>(x2, wck3, cb3, x3);
    conv_nhwc_kernel<128, 28, 28, 128, 14, 14>
        <<<dim3(13, 8, 64), 32, 0, stream>>>(x3, wck4, cb4, es);

    pool_kernel<<<64, 128, 0, stream>>>(es, pooled);
    gemm_ba_kernel<<<dim3(4, 8), 32, 0, stream>>>(pooled, 128, 128, nullptr, 0, 0,
                                                  wWg, bg, nullptr, envg, 128, 1);
    motion_kernel<<<64, 256, 0, stream>>>(history, Wm1, bm1, Wm2, bm2, mfeat);

    // --- history LSTM (60 steps, 2 layers) ---
    hipMemsetAsync(hh1[0], 0, 64 * 128 * 2, stream);
    hipMemsetAsync(hc1[0], 0, 64 * 128 * 4, stream);
    hipMemsetAsync(hh2[0], 0, 64 * 128 * 2, stream);
    hipMemsetAsync(hc2[0], 0, 64 * 128 * 4, stream);
    for (int t = 0; t < 60; ++t) {
        int p = t & 1, q = 1 - p;
        lstm_step_kernel<<<dim3(4, 8), 32, 0, stream>>>(hist16 + t * 32, 60 * 32, 32,
            whih0, hh1[p], whhh0, 128, hb0, hc1[p], hh1[q], hc1[q]);
        lstm_step_kernel<<<dim3(4, 8), 32, 0, stream>>>(hh1[q], 128, 128,
            whih1, hh2[p], whhh1, 128, hb1, hc2[p], hh2[q], hc2[q]);
    }
    const _Float16* hist_feat = hh2[0];  // last write: t=59 (p=1) -> index 0

    // context = relu([hist_feat | env_global] @ Wf^T + bf) + motion_feat
    gemm_ba_kernel<<<dim3(4, 16), 32, 0, stream>>>(hist_feat, 128, 128, envg, 128, 128,
                                                   wWf, bf, mfeat, context, 256, 1);

    // --- decoder init ---
    hipMemsetAsync(dh1[0], 0, 64 * 256 * 2, stream);
    hipMemsetAsync(dc1[0], 0, 64 * 256 * 4, stream);
    hipMemsetAsync(dc2[0], 0, 64 * 256 * 4, stream);
    hipMemsetAsync(posb[0], 0, 64 * 2 * 4, stream);
    hipMemcpyAsync(dh2[0], context, 64 * 256 * 2, hipMemcpyDeviceToDevice, stream);

    // --- decoder loop (360 steps) ---
    for (int t = 0; t < FUT; ++t) {
        int p = t & 1, q = 1 - p;
        dec_pre_kernel<<<64, 256, 0, stream>>>(es, posb[p], Ws, bs, Wp1, bp1, Wp2, bp2,
                                               context, scale_p, dec_in);
        lstm_step_kernel<<<dim3(4, 16), 32, 0, stream>>>(dec_in, 288, 288,
            wdWih0, dh1[p], wdWhh0, 256, db0, dc1[p], dh1[q], dc1[q]);
        lstm_step_kernel<<<dim3(4, 16), 32, 0, stream>>>(dh1[q], 256, 256,
            wdWih1, dh2[p], wdWhh1, 256, db1, dc2[p], dh2[q], dc2[q]);
        pos_out_kernel<<<1, 128, 0, stream>>>(dh2[q], Wo, bo, posb[p], posb[q],
                                              (float*)d_out, t);
    }
}
